// Struct2Seq_75376676044808
// MI455X (gfx1250) — matrix operand order repaired and checked by reading, not compiled, not run on hardware
//
#include <hip/hip_runtime.h>

// ---------------------------------------------------------------------------
// CDNA5 / gfx1250 Struct2Seq forward. wave32, v_wmma_f32_16x16x32_f16,
// async global->LDS staging for the gather-GEMM (ASYNCcnt path).
// ---------------------------------------------------------------------------

typedef __attribute__((ext_vector_type(16))) _Float16 v16h;
typedef __attribute__((ext_vector_type(8)))  float    v8f;

constexpr int CB  = 2;          // batch
constexpr int CN  = 1024;       // nodes
constexpr int CK  = 30;         // neighbors
constexpr int CH  = 256;        // hidden
constexpr int CEMB = 20;
constexpr int CBN = CB * CN;    // 2048
constexpr int CNE = CBN * CK;   // 61440 edge rows

union HFrag { v16h v; _Float16 h[16]; unsigned int u[8]; };

__device__ inline unsigned int pack2h(float a, float b) {
  HFrag t; t.h[0] = (_Float16)a; t.h[1] = (_Float16)b; return t.u[0];
}

// A fragment (16x32 f16 MxK) from packed-f16 LDS tile (row stride ldh halves).
// ISA 7.12.2 wave32: lanes 0-15 K in {0-7,16-23}, lanes 16-31 K in {8-15,24-31}.
__device__ inline v16h fragA_pk(const _Float16* base, int m, int ldh, int hi) {
  HFrag f;
#pragma unroll
  for (int p = 0; p < 8; ++p) {
    const int k = ((p & 4) ? 16 : 0) + (hi ? 8 : 0) + ((p & 3) << 1);
    f.u[p] = *(const unsigned int*)(base + m * ldh + k);
  }
  return f.v;
}

// A fragment from an f32 LDS tile with per-row scale (row stride ldf floats).
__device__ inline v16h fragA_f32(const float* base, int m, int ldf, int hi, float sc) {
  HFrag f;
#pragma unroll
  for (int p = 0; p < 8; ++p) {
    const int k = ((p & 4) ? 16 : 0) + (hi ? 8 : 0) + ((p & 3) << 1);
    const float2 v = *(const float2*)(base + m * ldf + k);
    f.h[2 * p]     = (_Float16)(v.x * sc);
    f.h[2 * p + 1] = (_Float16)(v.y * sc);
  }
  return f.v;
}

// B fragment (32x16 f16 KxN) from transposed packed LDS tile ldsB[col][k].
__device__ inline v16h fragB_pk(const _Float16* base, int col, int ldh, int hi) {
  HFrag f;
#pragma unroll
  for (int p = 0; p < 8; ++p) {
    const int k = ((p & 4) ? 16 : 0) + (hi ? 8 : 0) + ((p & 3) << 1);
    f.u[p] = *(const unsigned int*)(base + col * ldh + k);
  }
  return f.v;
}

// ---------------------------------------------------------------------------
// Node GEMM: C[M,N] = A[M,Kreal] @ W[Kreal,N] (+bias)(ReLU opt.)
// block 256 thr = 8 waves; macro-tile 64M x 128N; wave tile 16M x 64N (4 WMMA).
// ---------------------------------------------------------------------------
__global__ void gemm_node_wmma(const float* __restrict__ A, const float* __restrict__ W,
                               const float* __restrict__ bias, float* __restrict__ C,
                               int M, int Kreal, int Nreal, int relu) {
  __shared__ _Float16 ldsA[64][34];   // packed pairs along k
  __shared__ _Float16 ldsB[128][34];  // transposed: [col][k]
  const int t    = threadIdx.x;
  const int lane = t & 31;
  const int wave = t >> 5;
  const int mw   = wave >> 1;   // 0..3  (16-row sub-tile)
  const int nw   = wave & 1;    // 0..1  (64-col half)
  const int hi   = lane >> 4;
  const int row0 = blockIdx.x * 64;
  const int colb = blockIdx.y * 128;
  v8f acc[4] = {};

  for (int k0 = 0; k0 < Kreal; k0 += 32) {
    // stage A: 64 rows x 16 packed pairs
    for (int i = t; i < 64 * 16; i += 256) {
      const int r = i >> 4, p = i & 15;
      const int kg = k0 + 2 * p, rg = row0 + r;
      float x0 = 0.f, x1 = 0.f;
      if (rg < M) {
        if (kg < Kreal)     x0 = A[(size_t)rg * Kreal + kg];
        if (kg + 1 < Kreal) x1 = A[(size_t)rg * Kreal + kg + 1];
      }
      *(unsigned int*)&ldsA[r][2 * p] = pack2h(x0, x1);
    }
    // stage B transposed: 32 k x 128 col
    for (int i = t; i < 32 * 128; i += 256) {
      const int k = i >> 7, c = i & 127;
      const int kg = k0 + k, cg = colb + c;
      float x = 0.f;
      if (kg < Kreal && cg < Nreal) x = W[(size_t)kg * Nreal + cg];
      ldsB[c][k] = (_Float16)x;
    }
    if (k0 + 32 < Kreal && colb + (t & 127) < Nreal)
      __builtin_prefetch(&W[(size_t)(k0 + 32 + (t >> 7)) * Nreal + colb + (t & 127)], 0, 0);
    __syncthreads();
    const v16h a = fragA_pk(&ldsA[0][0], mw * 16 + (lane & 15), 34, hi);
#pragma unroll
    for (int f = 0; f < 4; ++f) {
      const v16h b = fragB_pk(&ldsB[0][0], nw * 64 + f * 16 + (lane & 15), 34, hi);
      acc[f] = __builtin_amdgcn_wmma_f32_16x16x32_f16(false, a, false, b, (short)0,
                                                      acc[f], false, false);
    }
    __syncthreads();
  }
#pragma unroll
  for (int f = 0; f < 4; ++f) {
    const int col = colb + nw * 64 + f * 16 + (lane & 15);
    if (col >= Nreal) continue;
    const float bv = bias ? bias[col] : 0.f;
#pragma unroll
    for (int j = 0; j < 8; ++j) {
      const int row = row0 + mw * 16 + j + hi * 8;
      if (row < M) {
        float v = acc[f][j] + bv;
        if (relu) v = fmaxf(v, 0.f);
        C[(size_t)row * Nreal + col] = v;
      }
    }
  }
}

// ---------------------------------------------------------------------------
// Fused gather/concat/mask edge GEMM, A-tile staged with async global->LDS DMA.
// Virtual A row e = concat of H-wide segments:
//   gather==0: base[e,:]    gather==1: base[b*CN + E_idx[e], :]
//   smode: 0 none / 1 *mask_bw[e] / 2 *2*mask_bw[e]   (applied at frag build)
// C[e,0:256] = Arow @ W[Kreal,256], Kreal in {512,768}. block 256, tile 64x128.
// ---------------------------------------------------------------------------
__global__ void gemm_edge_wmma(const float* __restrict__ s0, const float* __restrict__ s1,
                               const float* __restrict__ s2,
                               int g0, int g1, int g2, int m0, int m1, int m2,
                               const int* __restrict__ Eidx, const float* __restrict__ mbw,
                               const float* __restrict__ W, float* __restrict__ C,
                               int Kreal) {
  __shared__ float    ldsAe[64][36];   // f32 A tile (async DMA target), 16B-aligned rows
  __shared__ _Float16 ldsB[128][34];   // transposed packed weights
  __shared__ float    ldsScale[64];
  const int t    = threadIdx.x;
  const int lane = t & 31;
  const int wave = t >> 5;
  const int mw   = wave >> 1;
  const int nw   = wave & 1;
  const int hi   = lane >> 4;
  const int e0   = blockIdx.x * 64;
  const int colb = blockIdx.y * 128;
  // async staging geometry: 4 threads per row, 8 floats (2x b128) each
  const int sr = t >> 2;              // 0..63
  const int sc8 = (t & 3) << 3;       // 0,8,16,24
  if (t < 64) ldsScale[t] = mbw[e0 + t];
  v8f acc[4] = {};

  for (int k0 = 0; k0 < Kreal; k0 += 32) {
    const int seg    = k0 >> 8;
    const int klocal = k0 & 255;
    const float* base = (seg == 0) ? s0 : (seg == 1) ? s1 : s2;
    const int gmode   = (seg == 0) ? g0 : (seg == 1) ? g1 : g2;
    const int smode   = (seg == 0) ? m0 : (seg == 1) ? m1 : m2;
    {
      const int e = e0 + sr;
      int srow;
      if (gmode) srow = (e / (CN * CK)) * CN + Eidx[e];
      else       srow = e;
      const float* gp = base + (size_t)srow * CH + klocal + sc8;
      const unsigned int ldst = (unsigned int)(size_t)&ldsAe[sr][sc8];
      asm volatile("global_load_async_to_lds_b128 %0, %1, off\n\t"
                   "global_load_async_to_lds_b128 %0, %1, off offset:16"
                   :: "v"(ldst), "v"((unsigned long long)gp)
                   : "memory");
    }
    for (int i = t; i < 32 * 128; i += 256) {
      const int k = i >> 7, c = i & 127;
      ldsB[c][k] = (_Float16)W[(size_t)(k0 + k) * CH + colb + c];
    }
    if (k0 + 32 < Kreal)
      __builtin_prefetch(&W[(size_t)(k0 + 32 + (t >> 7)) * CH + colb + (t & 127)], 0, 0);
    asm volatile("s_wait_asynccnt 0x0" ::: "memory");
    __syncthreads();
    const int  mrow = mw * 16 + (lane & 15);
    const float mraw = ldsScale[mrow];
    const float sc = (smode == 0) ? 1.f : ((smode == 1) ? mraw : 2.f * mraw);
    const v16h a = fragA_f32(&ldsAe[0][0], mrow, 36, hi, sc);
#pragma unroll
    for (int f = 0; f < 4; ++f) {
      const v16h b = fragB_pk(&ldsB[0][0], nw * 64 + f * 16 + (lane & 15), 34, hi);
      acc[f] = __builtin_amdgcn_wmma_f32_16x16x32_f16(false, a, false, b, (short)0,
                                                      acc[f], false, false);
    }
    __syncthreads();
  }
#pragma unroll
  for (int f = 0; f < 4; ++f) {
    const int col = colb + nw * 64 + f * 16 + (lane & 15);
#pragma unroll
    for (int j = 0; j < 8; ++j) {
      const int row = e0 + mw * 16 + j + hi * 8;
      C[(size_t)row * CH + col] = acc[f][j];
    }
  }
}

// ---------------------------------------------------------------------------
// Neighbor attention: 1 block per (b,n), 1 wave32 per head (NH=4, d=64, K=30).
// ---------------------------------------------------------------------------
__global__ void attn_kernel(const float* __restrict__ Q, const float* __restrict__ Kp,
                            const float* __restrict__ Vp,
                            const float* __restrict__ mAtt, float* __restrict__ upd,
                            int use_mask) {
  __shared__ float att[4][32];
  const int bn   = blockIdx.x;
  const int h    = threadIdx.x >> 5;
  const int lane = threadIdx.x & 31;
  const float* q = Q + (size_t)bn * CH + h * 64;

  float logit = -1e30f;
  float mval  = 1.f;
  if (lane < CK) {
    const float* kr = Kp + ((size_t)bn * CK + lane) * CH + h * 64;
    float s = 0.f;
#pragma unroll 8
    for (int d = 0; d < 64; ++d) s += q[d] * kr[d];
    logit = s * 0.125f;  // 1/sqrt(64)
    if (use_mask) {
      mval  = mAtt[(size_t)bn * CK + lane];
      logit = (mval > 0.f) ? logit : -1e9f;
    }
  }
  float mx = logit;
#pragma unroll
  for (int o = 16; o > 0; o >>= 1) mx = fmaxf(mx, __shfl_xor(mx, o, 32));
  float e = (lane < CK) ? __expf(logit - mx) : 0.f;
  float sum = e;
#pragma unroll
  for (int o = 16; o > 0; o >>= 1) sum += __shfl_xor(sum, o, 32);
  float a = e / sum;
  if (use_mask) a *= mval;
  att[h][lane] = (lane < CK) ? a : 0.f;
  __syncthreads();

  float acc0 = 0.f, acc1 = 0.f;
  for (int k = 0; k < CK; ++k) {
    const float w = att[h][k];
    const float* vr = Vp + ((size_t)bn * CK + k) * CH + h * 64;
    acc0 += w * vr[lane];
    acc1 += w * vr[lane + 32];
  }
  upd[(size_t)bn * CH + h * 64 + lane]      = acc0;
  upd[(size_t)bn * CH + h * 64 + lane + 32] = acc1;
}

// ---------------------------------------------------------------------------
// out = maskV? * LayerNorm(x (+res)) * g + b     (1 block per row, 256 thr)
// ---------------------------------------------------------------------------
__global__ void add_ln_kernel(const float* __restrict__ x, const float* __restrict__ res,
                              const float* __restrict__ g, const float* __restrict__ bb,
                              const float* __restrict__ maskV, float* __restrict__ out) {
  __shared__ float s1[8], s2[8];
  const int row = blockIdx.x;
  const int t   = threadIdx.x;
  float v = x[(size_t)row * CH + t];
  if (res) v += res[(size_t)row * CH + t];
  float a = v, q = v * v;
#pragma unroll
  for (int o = 16; o > 0; o >>= 1) {
    a += __shfl_xor(a, o, 32);
    q += __shfl_xor(q, o, 32);
  }
  if ((t & 31) == 0) { s1[t >> 5] = a; s2[t >> 5] = q; }
  __syncthreads();
  if (t == 0) {
    float A = 0.f, Qq = 0.f;
    for (int i = 0; i < 8; ++i) { A += s1[i]; Qq += s2[i]; }
    s1[0] = A; s2[0] = Qq;
  }
  __syncthreads();
  const float mean = s1[0] * (1.f / CH);
  const float var  = s2[0] * (1.f / CH) - mean * mean;
  const float r    = rsqrtf(var + 1e-6f);
  float o = g[t] * (v - mean) * r + bb[t];
  if (maskV) o *= maskV[row];
  out[(size_t)row * CH + t] = o;
}

// mask_attend[e] = mask[b,n]*mask[b,Eidx[e]];  mask_bw[e] = mask[b,n]*(Eidx[e]!=n)
__global__ void mask_prep_kernel(const float* __restrict__ mask, const int* __restrict__ Eidx,
                                 float* __restrict__ mAtt, float* __restrict__ mBw) {
  const int e = blockIdx.x * 256 + threadIdx.x;
  if (e >= CNE) return;
  const int b = e / (CN * CK);
  const int n = (e / CK) % CN;
  const int j = Eidx[e];
  const float m = mask[b * CN + n];
  mAtt[e] = m * mask[b * CN + j];
  mBw[e]  = m * ((j != n) ? 1.f : 0.f);
}

__global__ void vec_copy_kernel(const float* __restrict__ a, float* __restrict__ o, int n) {
  const int i = blockIdx.x * 256 + threadIdx.x;
  if (i < n) o[i] = a[i];
}
__global__ void vec_add_kernel(const float* __restrict__ a, const float* __restrict__ b,
                               float* __restrict__ o, int n) {
  const int i = blockIdx.x * 256 + threadIdx.x;
  if (i < n) o[i] = a[i] + b[i];
}

// in-place: row of 20 logits -> sigmoid -> L2 normalize (1 wave per row)
__global__ void sig_norm_kernel(float* __restrict__ out) {
  const int row  = blockIdx.x;
  const int lane = threadIdx.x;
  float v = 0.f;
  if (lane < CEMB) v = 1.f / (1.f + __expf(-out[(size_t)row * CEMB + lane]));
  float s = v * v;
#pragma unroll
  for (int o = 16; o > 0; o >>= 1) s += __shfl_xor(s, o, 32);
  const float inv = rsqrtf(s);
  if (lane < CEMB) out[(size_t)row * CEMB + lane] = v * inv;
}

// ---------------------------------------------------------------------------
// Host-side orchestration
// ---------------------------------------------------------------------------
extern "C" void kernel_launch(void* const* d_in, const int* in_sizes, int n_in,
                              void* d_out, int out_size, void* d_ws, size_t ws_size,
                              hipStream_t stream) {
  (void)in_sizes; (void)n_in; (void)out_size; (void)ws_size;

  const float* V     = (const float*)d_in[0];
  const float* E     = (const float*)d_in[1];
  const float* S     = (const float*)d_in[2];
  const float* mask  = (const float*)d_in[3];
  const int*   Eidx  = (const int*)d_in[4];
  const float* Wv_w  = (const float*)d_in[5];
  const float* Wv_b  = (const float*)d_in[6];
  const float* We_w  = (const float*)d_in[7];
  const float* We_b  = (const float*)d_in[8];
  const float* Ws_w  = (const float*)d_in[9];
  const float* Ws_b  = (const float*)d_in[10];
  const float* P[2][12];  // enc/dec: WQ,WK,WV,WO,Wf1,bf1,Wf2,bf2,n1g,n1b,n2g,n2b
  for (int s = 0; s < 2; ++s)
    for (int j = 0; j < 12; ++j)
      P[s][j] = (const float*)d_in[11 + s * 12 + j];
  const float* Wo_w = (const float*)d_in[35];
  const float* Wo_b = (const float*)d_in[36];

  float* ws = (float*)d_ws;
  size_t off = 0;
  auto carve = [&](size_t n) { float* p = ws + off; off += (n + 63) & ~(size_t)63; return p; };
  float* hE    = carve((size_t)CNE * CH);
  float* Kp    = carve((size_t)CNE * CH);
  float* Vp    = carve((size_t)CNE * CH);
  float* hV    = carve((size_t)CBN * CH);
  float* hS    = carve((size_t)CBN * CH);
  float* hVenc = carve((size_t)CBN * CH);
  float* hVsum = carve((size_t)CBN * CH);
  float* Qb    = carve((size_t)CBN * CH);
  float* updB  = carve((size_t)CBN * CH);
  float* t1    = carve((size_t)CBN * CH);
  float* ht    = carve((size_t)CBN * CH);
  float* ffn   = carve((size_t)CBN * 4 * CH);
  float* mAtt  = carve(CNE);
  float* mBw   = carve(CNE);

  const dim3 blk(256);
  const dim3 gNode(CBN / 64, CH / 128);         // 32 x 2
  const dim3 gEdge(CNE / 64, CH / 128);         // 960 x 2
  const dim3 gEdgeInit(CNE / 64, CH / 128);
  const dim3 gFfn1(CBN / 64, (4 * CH) / 128);   // 32 x 8
  const dim3 gOut(CBN / 64, 1);                 // N=20 guard-padded

  mask_prep_kernel<<<(CNE + 255) / 256, 256, 0, stream>>>(mask, Eidx, mAtt, mBw);
  gemm_node_wmma<<<gNode, blk, 0, stream>>>(V, Wv_w, Wv_b, hV, CBN, 256, CH, 0);
  gemm_node_wmma<<<gEdgeInit, blk, 0, stream>>>(E, We_w, We_b, hE, CNE, 256, CH, 0);
  gemm_node_wmma<<<gNode, blk, 0, stream>>>(S, Ws_w, Ws_b, hS, CBN, CEMB, CH, 0);

  // encoder: h_EV = [h_E | gather(h_V)], Kd = 512
  for (int i = 0; i < 3; ++i) {
    const float* WQ = P[0][0] + (size_t)i * CH * CH;
    const float* WK = P[0][1] + (size_t)i * 2 * CH * CH;
    const float* WVw= P[0][2] + (size_t)i * 2 * CH * CH;
    const float* WO = P[0][3] + (size_t)i * CH * CH;
    const float* Wf1= P[0][4] + (size_t)i * CH * 4 * CH;
    const float* bf1= P[0][5] + (size_t)i * 4 * CH;
    const float* Wf2= P[0][6] + (size_t)i * 4 * CH * CH;
    const float* bf2= P[0][7] + (size_t)i * CH;
    const float* n1g= P[0][8] + (size_t)i * CH;
    const float* n1b= P[0][9] + (size_t)i * CH;
    const float* n2g= P[0][10] + (size_t)i * CH;
    const float* n2b= P[0][11] + (size_t)i * CH;

    gemm_edge_wmma<<<gEdge, blk, 0, stream>>>(hE, hV, nullptr, 0, 1, 0, 0, 0, 0,
                                              Eidx, mBw, WK, Kp, 512);
    gemm_edge_wmma<<<gEdge, blk, 0, stream>>>(hE, hV, nullptr, 0, 1, 0, 0, 0, 0,
                                              Eidx, mBw, WVw, Vp, 512);
    gemm_node_wmma<<<gNode, blk, 0, stream>>>(hV, WQ, nullptr, Qb, CBN, CH, CH, 0);
    attn_kernel<<<CBN, 128, 0, stream>>>(Qb, Kp, Vp, mAtt, updB, 1);
    gemm_node_wmma<<<gNode, blk, 0, stream>>>(updB, WO, nullptr, t1, CBN, CH, CH, 0);
    add_ln_kernel<<<CBN, 256, 0, stream>>>(t1, hV, n1g, n1b, nullptr, ht);
    gemm_node_wmma<<<gFfn1, blk, 0, stream>>>(ht, Wf1, bf1, ffn, CBN, CH, 4 * CH, 1);
    gemm_node_wmma<<<gNode, blk, 0, stream>>>(ffn, Wf2, bf2, t1, CBN, 4 * CH, CH, 0);
    add_ln_kernel<<<CBN, 256, 0, stream>>>(t1, ht, n2g, n2b, mask, hV);
  }
  vec_copy_kernel<<<(CBN * CH + 255) / 256, 256, 0, stream>>>(hV, hVenc, CBN * CH);

  // decoder: fused masked h_ESV, Kd = 768
  //   ch 0:H   -> 2*mask_bw * h_E
  //   ch H:2H  ->   mask_bw * gather(h_S)
  //   ch 2H:3H ->   mask_bw * gather(h_V + h_V_enc)
  for (int i = 0; i < 3; ++i) {
    const float* WQ = P[1][0] + (size_t)i * CH * CH;
    const float* WK = P[1][1] + (size_t)i * 3 * CH * CH;
    const float* WVw= P[1][2] + (size_t)i * 3 * CH * CH;
    const float* WO = P[1][3] + (size_t)i * CH * CH;
    const float* Wf1= P[1][4] + (size_t)i * CH * 4 * CH;
    const float* bf1= P[1][5] + (size_t)i * 4 * CH;
    const float* Wf2= P[1][6] + (size_t)i * 4 * CH * CH;
    const float* bf2= P[1][7] + (size_t)i * CH;
    const float* n1g= P[1][8] + (size_t)i * CH;
    const float* n1b= P[1][9] + (size_t)i * CH;
    const float* n2g= P[1][10] + (size_t)i * CH;
    const float* n2b= P[1][11] + (size_t)i * CH;

    vec_add_kernel<<<(CBN * CH + 255) / 256, 256, 0, stream>>>(hV, hVenc, hVsum, CBN * CH);
    gemm_edge_wmma<<<gEdge, blk, 0, stream>>>(hE, hS, hVsum, 0, 1, 1, 2, 1, 1,
                                              Eidx, mBw, WK, Kp, 768);
    gemm_edge_wmma<<<gEdge, blk, 0, stream>>>(hE, hS, hVsum, 0, 1, 1, 2, 1, 1,
                                              Eidx, mBw, WVw, Vp, 768);
    gemm_node_wmma<<<gNode, blk, 0, stream>>>(hV, WQ, nullptr, Qb, CBN, CH, CH, 0);
    attn_kernel<<<CBN, 128, 0, stream>>>(Qb, Kp, Vp, mAtt, updB, 0);
    gemm_node_wmma<<<gNode, blk, 0, stream>>>(updB, WO, nullptr, t1, CBN, CH, CH, 0);
    add_ln_kernel<<<CBN, 256, 0, stream>>>(t1, hV, n1g, n1b, nullptr, ht);
    gemm_node_wmma<<<gFfn1, blk, 0, stream>>>(ht, Wf1, bf1, ffn, CBN, CH, 4 * CH, 1);
    gemm_node_wmma<<<gNode, blk, 0, stream>>>(ffn, Wf2, bf2, t1, CBN, 4 * CH, CH, 0);
    add_ln_kernel<<<CBN, 256, 0, stream>>>(t1, ht, n2g, n2b, mask, hV);
  }

  gemm_node_wmma<<<gOut, blk, 0, stream>>>(hV, Wo_w, Wo_b, (float*)d_out,
                                           CBN, CH, CEMB, 0);
  sig_norm_kernel<<<CBN, 32, 0, stream>>>((float*)d_out);
}
